// MixtureOfExperts_45243185496830
// MI455X (gfx1250) — matrix operand order repaired and checked by reading, not compiled, not run on hardware
//
#include <hip/hip_runtime.h>
#include <hip/hip_bf16.h>

// ---------------------------------------------------------------------------
// MoE top-2 FFN for gfx1250 (MI455X). bf16 WMMA (v_wmma_f32_16x16x32_bf16),
// f32 accumulate, token compaction per expert, fused GEMM1+GELU+GEMM2.
// ---------------------------------------------------------------------------

#define BDIM   2
#define SDIM   2048
#define BS     (BDIM * SDIM)   // 4096 tokens
#define DDIM   1024
#define EDIM   8
#define DFF    4096
#define TM     32               // tokens per MoE tile
#define FC     512              // DFF chunk per fused stage
#define NCHUNK (DFF / FC)       // 8

typedef __attribute__((ext_vector_type(16))) __bf16 v16bf;
typedef __attribute__((ext_vector_type(8)))  __bf16 v8bf;
typedef __attribute__((ext_vector_type(8)))  float  v8f;

__device__ __forceinline__ __bf16 f2bf(float f) {
  union { float f; unsigned u; } a; a.f = f;
  unsigned u = a.u;
  unsigned r = (u + 0x7fffu + ((u >> 16) & 1u)) >> 16;   // round-to-nearest-even
  union { unsigned short s; __bf16 b; } o; o.s = (unsigned short)r;
  return o.b;
}

__device__ __forceinline__ v8f wmma_bf16(v16bf a, v16bf b, v8f c) {
  // D = A(16x32 bf16) * B(32x16 bf16) + C(16x16 f32)
  return __builtin_amdgcn_wmma_f32_16x16x32_bf16(false, a, false, b,
                                                 (short)0, c, false, false);
}

// A fragment (16x32, row-major source, row stride = `stride` elems).
// Lanes 0-15 row=lane, K=kk..kk+15 contiguous;
// lanes 16-31 same rows, K=kk+16..kk+31 (kb already includes the half offset).
__device__ __forceinline__ v16bf ld_fragA(const __bf16* rowp, int kb) {
  union { v16bf v; v8bf h[2]; } u;
  u.h[0] = *(const v8bf*)(rowp + kb);
  u.h[1] = *(const v8bf*)(rowp + kb + 8);
  return u.v;
}

// B fragment (32x16): lanes 0-15 col=lane, K=kk..kk+15 contiguous;
// lanes 16-31 col=lane-16, K=kk+16..kk+31.  `p` points at the lane's 16 K's.
__device__ __forceinline__ v16bf ld_fragB(const __bf16* p) {
  union { v16bf v; v8bf h[2]; } u;
  u.h[0] = *(const v8bf*)(p);
  u.h[1] = *(const v8bf*)(p + 8);
  return u.v;
}

// ---------------------------------------------------------------------------
__global__ void zero_cnt(int* cnt) {
  if (threadIdx.x < EDIM) cnt[threadIdx.x] = 0;
}

// src: [E][R][C] f32  ->  dst: [E][C][R] bf16   (LDS-tiled transpose)
__global__ void convert_transpose(const float* __restrict__ src,
                                  __bf16* __restrict__ dst, int R, int C) {
  __shared__ __bf16 tile[64][66];
  const int e  = blockIdx.z;
  const int c0 = blockIdx.x * 64;
  const int r0 = blockIdx.y * 64;
  const float* s = src + (size_t)e * R * C;
  __bf16*      d = dst + (size_t)e * R * C;
  const int tr = threadIdx.x >> 6;   // 0..3
  const int tc = threadIdx.x & 63;   // 0..63
#pragma unroll
  for (int p = 0; p < 16; ++p) {
    int r = p * 4 + tr;
    tile[r][tc] = f2bf(s[(size_t)(r0 + r) * C + c0 + tc]);
  }
  __syncthreads();
#pragma unroll
  for (int p = 0; p < 16; ++p) {
    int cc = p * 4 + tr;
    d[(size_t)(c0 + cc) * R + r0 + tc] = tile[tc][cc];
  }
}

// ---------------------------------------------------------------------------
// Gating: logits = x.Wg + bg ; top-2 (jax tie-break) ; softmax over the 2 ;
// compaction lists ; x -> bf16 ; out = w1*b2[e1] + w2*b2[e2].
__global__ void gate_kernel(const float* __restrict__ x,
                            const float* __restrict__ Wg,
                            const float* __restrict__ bg,
                            const float* __restrict__ b2,
                            __bf16* __restrict__ xb,
                            float* __restrict__ wbuf,
                            int* __restrict__ cnt,
                            int* __restrict__ route,
                            float* __restrict__ out) {
  const int t   = blockIdx.x;
  const int tid = threadIdx.x;           // 256 threads
  const int d0  = tid * 4;

  float acc[EDIM];
#pragma unroll
  for (int e = 0; e < EDIM; ++e) acc[e] = 0.f;

  const float4 x4 = *(const float4*)(x + (size_t)t * DDIM + d0);
  float xv[4] = {x4.x, x4.y, x4.z, x4.w};
#pragma unroll
  for (int j = 0; j < 4; ++j) {
    const float4* wp = (const float4*)(Wg + (size_t)(d0 + j) * EDIM);
    float4 a = wp[0], b = wp[1];
    acc[0] += xv[j] * a.x; acc[1] += xv[j] * a.y;
    acc[2] += xv[j] * a.z; acc[3] += xv[j] * a.w;
    acc[4] += xv[j] * b.x; acc[5] += xv[j] * b.y;
    acc[6] += xv[j] * b.z; acc[7] += xv[j] * b.w;
    xb[(size_t)t * DDIM + d0 + j] = f2bf(xv[j]);
  }

  // wave32 reduction
#pragma unroll
  for (int off = 16; off > 0; off >>= 1)
#pragma unroll
    for (int e = 0; e < EDIM; ++e) acc[e] += __shfl_down(acc[e], off, 32);

  __shared__ float red[8][EDIM];
  __shared__ float lg[EDIM];
  __shared__ int   sE[2];
  __shared__ float sW[2];
  const int wid = tid >> 5, lane = tid & 31;
  if (lane == 0)
#pragma unroll
    for (int e = 0; e < EDIM; ++e) red[wid][e] = acc[e];
  __syncthreads();
  if (tid < EDIM) {
    float s = bg[tid];
#pragma unroll
    for (int w = 0; w < 8; ++w) s += red[w][tid];
    lg[tid] = s;
  }
  __syncthreads();
  if (tid == 0) {
    int i1 = 0;
#pragma unroll
    for (int e = 1; e < EDIM; ++e) if (lg[e] > lg[i1]) i1 = e;   // first max wins
    int i2 = (i1 == 0) ? 1 : 0;
#pragma unroll
    for (int e = 0; e < EDIM; ++e)
      if (e != i1 && lg[e] > lg[i2]) i2 = e;
    const float l1 = lg[i1], l2 = lg[i2];
    const float p = __expf(l2 - l1);          // l1 >= l2, stable
    const float inv = 1.f / (1.f + p);
    const float w1 = inv, w2 = p * inv;
#pragma unroll
    for (int e = 0; e < EDIM; ++e) wbuf[t * EDIM + e] = 0.f;
    wbuf[t * EDIM + i1] = w1;
    wbuf[t * EDIM + i2] = w2;
    sE[0] = i1; sE[1] = i2; sW[0] = w1; sW[1] = w2;
    int p1 = atomicAdd(&cnt[i1], 1); route[i1 * BS + p1] = t;
    int p2 = atomicAdd(&cnt[i2], 1); route[i2 * BS + p2] = t;
  }
  __syncthreads();
  const float4 ba = *(const float4*)(b2 + (size_t)sE[0] * DDIM + d0);
  const float4 bb = *(const float4*)(b2 + (size_t)sE[1] * DDIM + d0);
  float4 o;
  o.x = sW[0] * ba.x + sW[1] * bb.x;
  o.y = sW[0] * ba.y + sW[1] * bb.y;
  o.z = sW[0] * ba.z + sW[1] * bb.z;
  o.w = sW[0] * ba.w + sW[1] * bb.w;
  *(float4*)(out + (size_t)t * DDIM + d0) = o;
}

// ---------------------------------------------------------------------------
// Fused expert FFN for one (expert, 32-token tile): h = gelu(x.W1 + b1);
// out += w * (h.W2).  16 waves; DFF processed in 8 chunks of 512 held in LDS.
__global__ __launch_bounds__(512) void moe_ffn(
    const __bf16* __restrict__ xb,     // [BS][D] bf16
    const __bf16* __restrict__ w1t,    // [E][DFF][D] bf16 (transposed)
    const __bf16* __restrict__ w2t,    // [E][D][DFF] bf16 (transposed)
    const float*  __restrict__ b1,     // [E][DFF]
    const float*  __restrict__ wbuf,   // [BS][E]
    const int*    __restrict__ cnt,    // [E]
    const int*    __restrict__ route,  // [E][BS]
    float* __restrict__ out) {         // [BS][D]
  const int e    = blockIdx.y;
  const int tile = blockIdx.x;
  const int n    = cnt[e];
  if (tile * TM >= n) return;

  __shared__ __align__(16) __bf16 xs[TM][DDIM];   // 64 KB
  __shared__ __align__(16) __bf16 hs[TM][FC];     // 32 KB
  __shared__ int   stok[TM];
  __shared__ float swt[TM];

  const int tid = threadIdx.x, wid = tid >> 5, lane = tid & 31;

  if (tid < TM) {
    const int g = tile * TM + tid;
    const bool valid = g < n;
    const int tok = route[e * BS + (valid ? g : tile * TM)];
    stok[tid] = tok;
    swt[tid]  = valid ? wbuf[tok * EDIM + e] : 0.f;
  }
  __syncthreads();

  { // gather 32 token rows of bf16 x into LDS (16 threads per row)
    const int row = tid >> 4, seg = tid & 15;
    const v8bf* sp = (const v8bf*)(xb + (size_t)stok[row] * DDIM + seg * 64);
    v8bf* dp = (v8bf*)(&xs[row][seg * 64]);
#pragma unroll
    for (int i = 0; i < 8; ++i) dp[i] = sp[i];
  }
  __syncthreads();

  const int hi   = (lane & 16) ? 1 : 0;       // second half-wave
  const int lrow = lane & 15;
  const __bf16* xrow0 = &xs[lrow][0];
  const __bf16* xrow1 = &xs[16 + lrow][0];
  const __bf16* hrow0 = &hs[lrow][0];
  const __bf16* hrow1 = &hs[16 + lrow][0];

  v8f acc[2][4];                               // 32x(wid's 64 cols of D)
#pragma unroll
  for (int m = 0; m < 2; ++m)
#pragma unroll
    for (int t2 = 0; t2 < 4; ++t2)
#pragma unroll
      for (int i = 0; i < 8; ++i) acc[m][t2][i] = 0.f;

  for (int ch = 0; ch < NCHUNK; ++ch) {
    const int fb = ch * FC;

    // ---- GEMM1: hacc[32 x 32cols] over K = D ----
    v8f hacc[2][2];
#pragma unroll
    for (int m = 0; m < 2; ++m)
#pragma unroll
      for (int t2 = 0; t2 < 2; ++t2)
#pragma unroll
        for (int i = 0; i < 8; ++i) hacc[m][t2][i] = 0.f;

    for (int kk = 0; kk < DDIM; kk += 32) {
      const int kbA = kk + hi * 16;
      const v16bf a0 = ld_fragA(xrow0, kbA);
      const v16bf a1 = ld_fragA(xrow1, kbA);
#pragma unroll
      for (int nt = 0; nt < 2; ++nt) {
        const int f = fb + wid * 32 + nt * 16 + lrow;
        const __bf16* bp =
            w1t + ((size_t)e * DFF + f) * DDIM + kk + hi * 16;
        const v16bf bfr = ld_fragB(bp);
        hacc[0][nt] = wmma_bf16(a0, bfr, hacc[0][nt]);
        hacc[1][nt] = wmma_bf16(a1, bfr, hacc[1][nt]);
      }
    }

    // bias + exact GELU -> bf16 h chunk in LDS
#pragma unroll
    for (int m = 0; m < 2; ++m)
#pragma unroll
      for (int nt = 0; nt < 2; ++nt) {
        const int cl = wid * 32 + nt * 16 + lrow;
        const float bias = b1[(size_t)e * DFF + fb + cl];
#pragma unroll
        for (int i = 0; i < 8; ++i) {
          const int r = m * 16 + i + hi * 8;
          const float v = hacc[m][nt][i] + bias;
          const float g = 0.5f * v * (1.f + erff(v * 0.70710678118654752f));
          hs[r][cl] = f2bf(g);
        }
      }
    __syncthreads();

    // ---- GEMM2 partial: acc += h_chunk . W2[fb:fb+FC, :] ----
    for (int kf = 0; kf < FC; kf += 32) {
      const int kbA = kf + hi * 16;
      const v16bf a0 = ld_fragA(hrow0, kbA);
      const v16bf a1 = ld_fragA(hrow1, kbA);
#pragma unroll
      for (int nt = 0; nt < 4; ++nt) {
        const int dcol = wid * 64 + nt * 16 + lrow;
        const __bf16* bp =
            w2t + ((size_t)e * DDIM + dcol) * DFF + fb + kf + hi * 16;
        const v16bf bfr = ld_fragB(bp);
        acc[0][nt] = wmma_bf16(a0, bfr, acc[0][nt]);
        acc[1][nt] = wmma_bf16(a1, bfr, acc[1][nt]);
      }
    }
    __syncthreads();   // protect hs before next chunk overwrites it
  }

  // ---- scale by routing weight, combine into out ----
#pragma unroll
  for (int m = 0; m < 2; ++m)
#pragma unroll
    for (int nt = 0; nt < 4; ++nt) {
      const int col = wid * 64 + nt * 16 + lrow;
#pragma unroll
      for (int i = 0; i < 8; ++i) {
        const int r = m * 16 + i + hi * 8;
        unsafeAtomicAdd(out + (size_t)stok[r] * DDIM + col,
                        swt[r] * acc[m][nt][i]);
      }
    }
}

// ---------------------------------------------------------------------------
extern "C" void kernel_launch(void* const* d_in, const int* in_sizes, int n_in,
                              void* d_out, int out_size, void* d_ws, size_t ws_size,
                              hipStream_t stream) {
  const float* x  = (const float*)d_in[0];
  const float* W1 = (const float*)d_in[1];
  const float* b1 = (const float*)d_in[2];
  const float* W2 = (const float*)d_in[3];
  const float* b2 = (const float*)d_in[4];
  const float* Wg = (const float*)d_in[5];
  const float* bg = (const float*)d_in[6];
  float* out = (float*)d_out;

  char* p = (char*)d_ws;
  int* cnt = (int*)p;                 p += 256;
  int* route = (int*)p;               p += (size_t)EDIM * BS * sizeof(int);
  float* wbuf = (float*)p;            p += (size_t)BS * EDIM * sizeof(float);
  __bf16* xb  = (__bf16*)p;           p += (size_t)BS * DDIM * 2;
  __bf16* w1t = (__bf16*)p;           p += (size_t)EDIM * DFF * DDIM * 2;
  __bf16* w2t = (__bf16*)p;           p += (size_t)EDIM * DDIM * DFF * 2;

  zero_cnt<<<1, 32, 0, stream>>>(cnt);
  // W1 [E][D][DFF] -> [E][DFF][D]
  convert_transpose<<<dim3(DFF / 64, DDIM / 64, EDIM), 256, 0, stream>>>(
      W1, w1t, DDIM, DFF);
  // W2 [E][DFF][D] -> [E][D][DFF]
  convert_transpose<<<dim3(DDIM / 64, DFF / 64, EDIM), 256, 0, stream>>>(
      W2, w2t, DFF, DDIM);
  gate_kernel<<<BS, 256, 0, stream>>>(x, Wg, bg, b2, xb, wbuf, cnt, route, out);
  moe_ffn<<<dim3(BS / TM, EDIM), 512, 0, stream>>>(
      xb, w1t, w2t, b1, wbuf, cnt, route, out);
}
